// GatedLinearAttention_84885733638208
// MI455X (gfx1250) — matrix operand order repaired and checked by reading, not compiled, not run on hardware
//
#include <hip/hip_runtime.h>

// ---------------------------------------------------------------------------
// GLA forward for MI455X (gfx1250, wave32, WMMA).
// Chunked gated-linear-attention: all heavy math on v_wmma_f32_16x16x32_bf16,
// GEMM staging via async global->LDS copies (ASYNCcnt) when available.
// ---------------------------------------------------------------------------

typedef __attribute__((ext_vector_type(16))) __bf16 v16bf;
typedef __attribute__((ext_vector_type(8)))  float  v8f;
typedef __attribute__((ext_vector_type(4)))  int    v4i;

#define T_TOK   8192
#define HID     2048
#define NHEADS  16
#define NKVH    4
#define HD      128
#define RLOW    16
#define QKV_N   ((NHEADS + 2*NKVH)*HD)   // 3072
#define CHUNK   64
#define NCHUNK  (T_TOK / CHUNK)          // 128
#define GDIM    (NKVH*HD)                // 512
#define QCOLS   (NHEADS*HD)              // 2048
#define KENDCOL (QCOLS + NKVH*HD)        // 2560
#define SCALE_Q 0.08838834764831845f     // 128^-0.5
#define INV_NORM_G (1.0f/16.0f)
#define EPS_R   1e-6f

// ---------------- CDNA5 async global->LDS path (guarded) --------------------
#if defined(__has_builtin)
#  if __has_builtin(__builtin_amdgcn_global_load_async_to_lds_b128)
#    define HAVE_ASYNC_B128 1
#  endif
#  if __has_builtin(__builtin_amdgcn_s_wait_asynccnt)
#    define HAVE_WAIT_ASYNC 1
#  endif
#endif

#ifdef HAVE_ASYNC_B128
typedef __attribute__((address_space(1))) v4i gv4i;   // global (device) int4
typedef __attribute__((address_space(3))) v4i lv4i;   // LDS int4
__device__ __forceinline__ void async_copy16(const __bf16* g, __bf16* l) {
  // 16 bytes per lane, memory -> LDS, tracked by ASYNCcnt
  __builtin_amdgcn_global_load_async_to_lds_b128((gv4i*)g, (lv4i*)l, 0, 0);
}
__device__ __forceinline__ void wait_async0() {
#ifdef HAVE_WAIT_ASYNC
  __builtin_amdgcn_s_wait_asynccnt(0);
#else
  asm volatile("s_wait_asynccnt 0x0" ::: "memory");
#endif
}
#endif

// ---------------- WMMA helpers (wave32; A/B both use k = k0 + 16*(lane/16) + s) -----

__device__ __forceinline__ v8f v8f_zero() {
  v8f z;
#pragma unroll
  for (int i = 0; i < 8; ++i) z[i] = 0.0f;
  return z;
}

// A fragment (16x32 bf16): lane<16 -> row m0+lane, K = k0+{0..15}
//                          lane>=16 -> row m0+lane-16, K = k0+{16..31}
__device__ __forceinline__ v16bf ldsA_frag(const __bf16* base, int m0, int k0,
                                           int ld, int lane) {
  const int r  = m0 + (lane & 15);
  const int kb = k0 + ((lane & 16) ? 16 : 0);
  const __bf16* p = base + r * ld + kb;
  v16bf a;
#pragma unroll
  for (int i = 0; i < 8; ++i) a[i] = p[i];
#pragma unroll
  for (int i = 0; i < 8; ++i) a[8 + i] = p[8 + i];
  return a;
}

// B fragment (32x16 bf16) from LDS tile holding B^T row-major.
__device__ __forceinline__ v16bf ldsB_frag(const __bf16* baseT, int n0, int k0,
                                           int ld, int lane) {
  const int n  = n0 + (lane & 15);
  const int kb = k0 + ((lane & 16) ? 16 : 0);
  const __bf16* p = baseT + n * ld + kb;
  v16bf b;
#pragma unroll
  for (int i = 0; i < 16; ++i) b[i] = p[i];
  return b;
}

__device__ __forceinline__ v8f wmma_bf16(v16bf a, v16bf b, v8f c) {
  return __builtin_amdgcn_wmma_f32_16x16x32_bf16(false, a, false, b,
                                                 (short)0, c, false, false);
}

// ---------------- conversion kernels ---------------------------------------

__global__ void conv_bf16_kernel(const float* __restrict__ src,
                                 __bf16* __restrict__ dst, size_t n) {
  size_t i = (size_t)blockIdx.x * 256 + threadIdx.x;
  if (i < n) dst[i] = (__bf16)src[i];
}

// dst[c][r] = src[r][c]  (pre-transpose weights so GEMM B staging is coalesced)
__global__ void convT_bf16_kernel(const float* __restrict__ src,
                                  __bf16* __restrict__ dst, int R, int Cc) {
  size_t i = (size_t)blockIdx.x * 256 + threadIdx.x;
  if (i < (size_t)R * Cc) {
    int r = (int)(i / Cc);
    int c = (int)(i % Cc);
    dst[(size_t)c * R + r] = (__bf16)src[i];
  }
}

// ---------------- generic bf16 WMMA GEMM: C = A(bf16) * B + bias -----------
// A: [M,K] row-major bf16.  BT: [N,K] row-major bf16 (B transposed).
// Double-buffered LDS pipeline; async global->LDS copies when available.
// LDS layout (elements): buf b: As at b*8192, Bs at b*8192 + 4096.
// MODE 0: plain f32 out.  MODE 1: qkv epilogue (relu+scale q, relu k).
template <int MODE>
__global__ void gemm_bf16_kernel(const __bf16* __restrict__ A,
                                 const __bf16* __restrict__ BT,
                                 const float* __restrict__ bias,
                                 float* __restrict__ C,
                                 int M, int N, int K) {
  extern __shared__ char smem[];
  __bf16* lds = (__bf16*)smem;           // 2 x (4096 A + 4096 B) elements

  const int tid  = threadIdx.x;
  const int lane = tid & 31;
  const int w    = tid >> 5;
  const int wm   = w >> 1;               // 0..3 -> M
  const int wn   = w & 1;                // 0..1 -> N
  const int mBase = blockIdx.y * 128;
  const int nBase = blockIdx.x * 128;

  v8f acc[2][4];
#pragma unroll
  for (int tm = 0; tm < 2; ++tm)
#pragma unroll
    for (int tn = 0; tn < 4; ++tn) acc[tm][tn] = v8f_zero();

  const int row  = tid >> 1;             // 0..127
  const int part = (tid & 1) * 16;       // 0 or 16 elements (32B halves)
  const __bf16* gA = A  + (size_t)(mBase + row) * K + part;
  const __bf16* gB = BT + (size_t)(nBase + row) * K + part;
  const int ldst = row * 32 + part;
  const int nsteps = K >> 5;

  // -------- prologue: stage step 0 into buffer 0 --------
#ifdef HAVE_ASYNC_B128
  async_copy16(gA,     lds + ldst);
  async_copy16(gA + 8, lds + ldst + 8);
  async_copy16(gB,     lds + 4096 + ldst);
  async_copy16(gB + 8, lds + 4096 + ldst + 8);
  wait_async0();
#else
  {
    uint4 a0 = ((const uint4*)gA)[0], a1 = ((const uint4*)gA)[1];
    uint4 b0 = ((const uint4*)gB)[0], b1 = ((const uint4*)gB)[1];
    ((uint4*)(lds + ldst))[0] = a0;        ((uint4*)(lds + ldst))[1] = a1;
    ((uint4*)(lds + 4096 + ldst))[0] = b0; ((uint4*)(lds + 4096 + ldst))[1] = b1;
  }
#endif
  __syncthreads();

  for (int s = 0; s < nsteps; ++s) {
    const int cur = s & 1;
    const int nxt = cur ^ 1;
    __bf16* curBuf = lds + cur * 8192;
    __bf16* nxtBuf = lds + nxt * 8192;
    const __bf16* nA = gA + (size_t)(s + 1) * 32;
    const __bf16* nB = gB + (size_t)(s + 1) * 32;

#ifdef HAVE_ASYNC_B128
    if (s + 1 < nsteps) {                // overlap copy of tile s+1 with math
      async_copy16(nA,     nxtBuf + ldst);
      async_copy16(nA + 8, nxtBuf + ldst + 8);
      async_copy16(nB,     nxtBuf + 4096 + ldst);
      async_copy16(nB + 8, nxtBuf + 4096 + ldst + 8);
      if (s + 2 < nsteps) {
        __builtin_prefetch(nA + 32, 0, 1);
        __builtin_prefetch(nB + 32, 0, 1);
      }
    }
#else
    uint4 a0, a1, b0, b1;
    if (s + 1 < nsteps) {                // register prefetch of tile s+1
      a0 = ((const uint4*)nA)[0]; a1 = ((const uint4*)nA)[1];
      b0 = ((const uint4*)nB)[0]; b1 = ((const uint4*)nB)[1];
      if (s + 2 < nsteps) {
        __builtin_prefetch(nA + 32, 0, 1);
        __builtin_prefetch(nB + 32, 0, 1);
      }
    }
#endif

    // -------- math on buffer `cur` --------
    const __bf16* As = curBuf;
    const __bf16* Bs = curBuf + 4096;
    v16bf aF[2];
#pragma unroll
    for (int tm = 0; tm < 2; ++tm)
      aF[tm] = ldsA_frag(As, wm * 32 + tm * 16, 0, 32, lane);
#pragma unroll
    for (int tn = 0; tn < 4; ++tn) {
      v16bf bF = ldsB_frag(Bs, wn * 64 + tn * 16, 0, 32, lane);
      acc[0][tn] = wmma_bf16(aF[0], bF, acc[0][tn]);
      acc[1][tn] = wmma_bf16(aF[1], bF, acc[1][tn]);
    }

#ifdef HAVE_ASYNC_B128
    wait_async0();
    __syncthreads();
#else
    if (s + 1 < nsteps) {
      ((uint4*)(nxtBuf + ldst))[0] = a0;        ((uint4*)(nxtBuf + ldst))[1] = a1;
      ((uint4*)(nxtBuf + 4096 + ldst))[0] = b0; ((uint4*)(nxtBuf + 4096 + ldst))[1] = b1;
    }
    __syncthreads();
#endif
  }

  // epilogue (C/D layout: N = lane&15, M = vgpr index (+8 for upper lanes))
  const int nLoc = lane & 15;
  const int mOff = (lane & 16) ? 8 : 0;
#pragma unroll
  for (int tm = 0; tm < 2; ++tm) {
#pragma unroll
    for (int tn = 0; tn < 4; ++tn) {
      const int gn = nBase + wn * 64 + tn * 16 + nLoc;
      const float bv = (MODE == 1) ? bias[gn] : 0.0f;
#pragma unroll
      for (int r = 0; r < 8; ++r) {
        const int gm = mBase + wm * 32 + tm * 16 + mOff + r;
        float v = acc[tm][tn][r] + bv;
        if (MODE == 1) {
          if (gn < QCOLS)        v = fmaxf(v, 0.0f) * SCALE_Q;  // q: relu*scale
          else if (gn < KENDCOL) v = fmaxf(v, 0.0f);            // k: relu
        }
        C[(size_t)gm * N + gn] = v;
      }
    }
  }
}

// ---------------- low-rank gate: g = log_sigmoid(hs@w0@w1 + b1)/16 ---------
__global__ void gate_kernel(const float* __restrict__ hs,
                            const float* __restrict__ w0,
                            const float* __restrict__ w1,
                            const float* __restrict__ b1,
                            float* __restrict__ g) {
  __shared__ float red[RLOW * 256];
  __shared__ float t16[RLOW];
  const int t = blockIdx.x, tid = threadIdx.x;

  float acc[RLOW];
#pragma unroll
  for (int r = 0; r < RLOW; ++r) acc[r] = 0.0f;
  for (int i = 0; i < HID / 256; ++i) {
    const int idx = tid + i * 256;
    const float x = hs[(size_t)t * HID + idx];
    const float* wr = w0 + (size_t)idx * RLOW;
#pragma unroll
    for (int r = 0; r < RLOW; ++r) acc[r] += x * wr[r];
  }
#pragma unroll
  for (int r = 0; r < RLOW; ++r) red[r * 256 + tid] = acc[r];
  __syncthreads();
  for (int off = 128; off > 0; off >>= 1) {
    if (tid < off)
#pragma unroll
      for (int r = 0; r < RLOW; ++r) red[r * 256 + tid] += red[r * 256 + tid + off];
    __syncthreads();
  }
  if (tid < RLOW) t16[tid] = red[tid * 256];
  __syncthreads();

  for (int c = tid; c < GDIM; c += 256) {
    float v = b1[c];
#pragma unroll
    for (int r = 0; r < RLOW; ++r) v += t16[r] * w1[r * GDIM + c];
    const float m  = fminf(v, 0.0f);
    const float ls = m - log1pf(expf(-fabsf(v)));          // stable log_sigmoid
    g[(size_t)t * GDIM + c] = ls * INV_NORM_G;
  }
}

// ---------------- per-(kv,chunk) prep: cumsum gates, gated q/k, bf16 v -----
__global__ void prep_kernel(const float* __restrict__ qkv,
                            const float* __restrict__ g,
                            float* __restrict__ expB,
                            __bf16* __restrict__ qg,  __bf16* __restrict__ kg,
                            __bf16* __restrict__ kBo, __bf16* __restrict__ vb) {
  __shared__ float gb[CHUNK * HD];                          // 32 KB
  const int c = blockIdx.x, kv = blockIdx.y, tid = threadIdx.x;
  const int t0 = c * CHUNK;

  for (int i = tid; i < CHUNK * HD; i += 256) {
    const int j = i >> 7, d = i & 127;
    gb[i] = g[(size_t)(t0 + j) * GDIM + kv * HD + d];
  }
  __syncthreads();
  if (tid < HD) {                                          // inclusive cumsum per d
    float run = 0.0f;
    for (int j = 0; j < CHUNK; ++j) { run += gb[j * HD + tid]; gb[j * HD + tid] = run; }
    expB[(kv * NCHUNK + c) * HD + tid] = expf(run);
  }
  __syncthreads();

  for (int i = tid; i < CHUNK * HD; i += 256) {
    const int j = i >> 7, d = i & 127;
    const int tok = t0 + j;
    const float b = gb[i];
    const float B = gb[(CHUNK - 1) * HD + d];
    const size_t qrow = (size_t)tok * QKV_N;
    const float kvk = qkv[qrow + QCOLS + kv * HD + d];      // relu already applied
    const float vv  = qkv[qrow + KENDCOL + kv * HD + d];
    const size_t o5 = (size_t)tok * GDIM + kv * HD + d;
    kg[o5]  = (__bf16)(kvk * expf(-b));
    kBo[o5] = (__bf16)(kvk * expf(B - b));
    vb[o5]  = (__bf16)vv;
    const float eb = expf(b);
#pragma unroll
    for (int r = 0; r < NHEADS / NKVH; ++r) {
      const int h = kv * (NHEADS / NKVH) + r;               // repeat_interleave GQA
      const float qv = qkv[qrow + h * HD + d];              // relu+scale applied
      qg[(size_t)tok * QCOLS + h * HD + d] = (__bf16)(qv * eb);
    }
  }
}

// ---------------- per-chunk state delta: dS = (k*e^{B-b})^T @ V ------------
__global__ void dstate_kernel(const __bf16* __restrict__ kBo,
                              const __bf16* __restrict__ vb,
                              float* __restrict__ S) {
  extern __shared__ char smem[];
  __bf16* Akt = (__bf16*)smem;            // [128(d)][64(j)]  (kB^T)
  __bf16* BvT = (__bf16*)(smem + 16384);  // [128(e)][64(j)]  (V^T, as B^T)
  const int c = blockIdx.x, kv = blockIdx.y, tid = threadIdx.x;
  const int t0 = c * CHUNK;

  for (int i = tid; i < CHUNK * HD; i += 256) {
    const int j = i >> 7, d = i & 127;
    const size_t o5 = (size_t)(t0 + j) * GDIM + kv * HD + d;
    Akt[d * CHUNK + j] = kBo[o5];
    BvT[d * CHUNK + j] = vb[o5];
  }
  __syncthreads();

  const int lane = tid & 31, w = tid >> 5, wm = w >> 1, wn = w & 1;
  v8f acc[2][4];
#pragma unroll
  for (int tm = 0; tm < 2; ++tm)
#pragma unroll
    for (int tn = 0; tn < 4; ++tn) acc[tm][tn] = v8f_zero();

#pragma unroll
  for (int k0 = 0; k0 < CHUNK; k0 += 32) {
    v16bf aF[2];
#pragma unroll
    for (int tm = 0; tm < 2; ++tm)
      aF[tm] = ldsA_frag(Akt, wm * 32 + tm * 16, k0, CHUNK, lane);
#pragma unroll
    for (int tn = 0; tn < 4; ++tn) {
      v16bf bF = ldsB_frag(BvT, wn * 64 + tn * 16, k0, CHUNK, lane);
      acc[0][tn] = wmma_bf16(aF[0], bF, acc[0][tn]);
      acc[1][tn] = wmma_bf16(aF[1], bF, acc[1][tn]);
    }
  }

  float* Sc = S + ((size_t)(kv * NCHUNK + c) << 14);
  const int nLoc = lane & 15, mOff = (lane & 16) ? 8 : 0;
#pragma unroll
  for (int tm = 0; tm < 2; ++tm)
#pragma unroll
    for (int tn = 0; tn < 4; ++tn) {
      const int gn = wn * 64 + tn * 16 + nLoc;
#pragma unroll
      for (int r = 0; r < 8; ++r) {
        const int gm = wm * 32 + tm * 16 + mOff + r;
        Sc[(size_t)gm * HD + gn] = acc[tm][tn][r];
      }
    }
}

// ---------------- in-place exclusive gated scan over chunks ----------------
// One thread per (kv, d, e): run' = expB[c][d]*run + dS[c];  S[c] <- run.
__global__ void scan_kernel(float* __restrict__ S, const float* __restrict__ expB) {
  const int gid = blockIdx.x * 256 + threadIdx.x;   // 0..65535
  const int kv = gid >> 14;
  const int de = gid & 16383;
  const int d  = de >> 7;
  float run = 0.0f;
  for (int c = 0; c < NCHUNK; ++c) {
    const size_t idx = ((size_t)(kv * NCHUNK + c) << 14) + de;
    const float delta = S[idx];
    S[idx] = run;
    run = expB[(kv * NCHUNK + c) * HD + d] * run + delta;
  }
}

// ---------------- per-(head,chunk) output + fused RMSNorm ------------------
__global__ void output_kernel(const __bf16* __restrict__ qg,
                              const __bf16* __restrict__ kg,
                              const __bf16* __restrict__ vb,
                              const float* __restrict__ S,
                              const float* __restrict__ gnorm,
                              __bf16* __restrict__ onorm) {
  extern __shared__ char smem[];
  __bf16* qgS = (__bf16*)smem;                        // [64][128] 16KB
  __bf16* kgS = (__bf16*)(smem + 16384);              // [64][128] 16KB
  __bf16* AmS = kgS;                                  // reuse: [64][64] 8KB
  float*  scaleRow = (float*)(smem + 16384 + 8192);   // 64 floats (spare)
  __bf16* vbT = (__bf16*)(smem + 32768);              // [128(e)][64(j)] 16KB
  __bf16* SbT = (__bf16*)(smem + 49152);              // [128(e)][128(d)] 32KB
  float*  oS  = (float*)(smem + 49152);               // reuse: [64][128] f32

  const int c = blockIdx.x, h = blockIdx.y, tid = threadIdx.x;
  const int kv = h >> 2;
  const int t0 = c * CHUNK;

  // stage qg / kg tiles (contiguous rows -> async-capable)
#ifdef HAVE_ASYNC_B128
  {
    const __bf16* gq = qg + (size_t)t0 * QCOLS + h * HD;
    const __bf16* gk = kg + (size_t)t0 * GDIM + kv * HD;
    for (int i = tid * 8; i < CHUNK * HD; i += 256 * 8) {
      const int j = i >> 7, d = i & 127;
      async_copy16(gq + (size_t)j * QCOLS + d, qgS + i);
      async_copy16(gk + (size_t)j * GDIM + d, kgS + i);
    }
    wait_async0();
  }
#else
  for (int i = tid; i < CHUNK * HD; i += 256) {
    const int j = i >> 7, d = i & 127;
    qgS[i] = qg[(size_t)(t0 + j) * QCOLS + h * HD + d];
    kgS[i] = kg[(size_t)(t0 + j) * GDIM + kv * HD + d];
  }
#endif
  __syncthreads();

  const int lane = tid & 31, w = tid >> 5, wm = w >> 1, wn = w & 1;
  const int nLoc = lane & 15, mOff = (lane & 16) ? 8 : 0;

  // ---- intra-chunk attention A = Qg @ Kg^T  (M=64,N=64,K=128) ----
  v8f accA[2] = { v8f_zero(), v8f_zero() };
#pragma unroll
  for (int k0 = 0; k0 < HD; k0 += 32) {
    v16bf aF = ldsA_frag(qgS, wm * 16, k0, HD, lane);
#pragma unroll
    for (int tn = 0; tn < 2; ++tn) {
      v16bf bF = ldsB_frag(kgS, wn * 32 + tn * 16, k0, HD, lane);
      accA[tn] = wmma_bf16(aF, bF, accA[tn]);
    }
  }
  __syncthreads();   // done reading kgS before overwriting with AmS

  // causal mask (keep j<=i) + store bf16 A
#pragma unroll
  for (int tn = 0; tn < 2; ++tn) {
    const int n = wn * 32 + tn * 16 + nLoc;
#pragma unroll
    for (int r = 0; r < 8; ++r) {
      const int m = wm * 16 + mOff + r;
      AmS[m * CHUNK + n] = (__bf16)((n <= m) ? accA[tn][r] : 0.0f);
    }
  }
  // stage V^T and S_start^T (bf16, transposed -> scalar staging)
  for (int i = tid; i < CHUNK * HD; i += 256) {
    const int j = i >> 7, e = i & 127;
    vbT[e * CHUNK + j] = vb[(size_t)(t0 + j) * GDIM + kv * HD + e];
  }
  const float* Sc = S + ((size_t)(kv * NCHUNK + c) << 14);
  for (int i = tid; i < HD * HD; i += 256) {
    const int d = i >> 7, e = i & 127;
    SbT[e * HD + d] = (__bf16)Sc[i];
  }
  __syncthreads();

  // ---- o = Amask @ V + Qg @ S_start   (M=64, N=128) ----
  v8f accO[4];
#pragma unroll
  for (int tn = 0; tn < 4; ++tn) accO[tn] = v8f_zero();
  const int m0 = wm * 16;
#pragma unroll
  for (int k0 = 0; k0 < CHUNK; k0 += 32) {
    v16bf aF = ldsA_frag(AmS, m0, k0, CHUNK, lane);
#pragma unroll
    for (int tn = 0; tn < 4; ++tn) {
      v16bf bF = ldsB_frag(vbT, wn * 64 + tn * 16, k0, CHUNK, lane);
      accO[tn] = wmma_bf16(aF, bF, accO[tn]);
    }
  }
#pragma unroll
  for (int k0 = 0; k0 < HD; k0 += 32) {
    v16bf aF = ldsA_frag(qgS, m0, k0, HD, lane);
#pragma unroll
    for (int tn = 0; tn < 4; ++tn) {
      v16bf bF = ldsB_frag(SbT, wn * 64 + tn * 16, k0, HD, lane);
      accO[tn] = wmma_bf16(aF, bF, accO[tn]);
    }
  }
  __syncthreads();   // done reading SbT, reuse as f32 o tile

#pragma unroll
  for (int tn = 0; tn < 4; ++tn) {
    const int n = wn * 64 + tn * 16 + nLoc;
#pragma unroll
    for (int r = 0; r < 8; ++r) oS[(m0 + mOff + r) * HD + n] = accO[tn][r];
  }
  __syncthreads();

  // fused per-row RMSNorm over head_dim
  if (tid < CHUNK) {
    float ss = 0.0f;
#pragma unroll 4
    for (int e = 0; e < HD; ++e) { const float x = oS[tid * HD + e]; ss += x * x; }
    scaleRow[tid] = rsqrtf(ss * (1.0f / HD) + EPS_R);
  }
  __syncthreads();
  for (int i = tid; i < CHUNK * HD; i += 256) {
    const int j = i >> 7, e = i & 127;
    const float val = oS[i] * scaleRow[j] * gnorm[e];
    onorm[(size_t)(t0 + j) * QCOLS + h * HD + e] = (__bf16)val;
  }
}

// ---------------------------------------------------------------------------

extern "C" void kernel_launch(void* const* d_in, const int* in_sizes, int n_in,
                              void* d_out, int out_size, void* d_ws, size_t ws_size,
                              hipStream_t stream) {
  (void)in_sizes; (void)n_in; (void)out_size; (void)ws_size;
  const float* hs    = (const float*)d_in[0];
  const float* Wqkv  = (const float*)d_in[1];
  const float* bqkv  = (const float*)d_in[2];
  const float* gk_w0 = (const float*)d_in[3];
  const float* gk_w1 = (const float*)d_in[4];
  const float* gk_b1 = (const float*)d_in[5];
  const float* gnorm = (const float*)d_in[6];
  const float* Wo    = (const float*)d_in[7];
  float* out = (float*)d_out;

  char* ws = (char*)d_ws;
  size_t off = 0;
  auto carve = [&](size_t bytes) -> void* {
    void* p = ws + off;
    off += (bytes + 255) & ~(size_t)255;
    return p;
  };
  __bf16* hsb   = (__bf16*)carve((size_t)T_TOK * HID * 2);
  __bf16* wqkvT = (__bf16*)carve((size_t)QKV_N * HID * 2);
  __bf16* woT   = (__bf16*)carve((size_t)HID * QCOLS * 2);
  float*  qkv   = (float*) carve((size_t)T_TOK * QKV_N * 4);
  float*  g     = (float*) carve((size_t)T_TOK * GDIM * 4);
  __bf16* qg    = (__bf16*)carve((size_t)T_TOK * QCOLS * 2);
  __bf16* kg    = (__bf16*)carve((size_t)T_TOK * GDIM * 2);
  __bf16* kBo   = (__bf16*)carve((size_t)T_TOK * GDIM * 2);
  __bf16* vb    = (__bf16*)carve((size_t)T_TOK * GDIM * 2);
  float*  expB  = (float*) carve((size_t)NKVH * NCHUNK * HD * 4);
  float*  S     = (float*) carve((size_t)NKVH * NCHUNK * HD * HD * 4);
  __bf16* onorm = (__bf16*)carve((size_t)T_TOK * QCOLS * 2);

  const size_t nhs = (size_t)T_TOK * HID;
  conv_bf16_kernel<<<dim3((unsigned)((nhs + 255) / 256)), 256, 0, stream>>>(hs, hsb, nhs);
  convT_bf16_kernel<<<dim3((unsigned)(((size_t)HID * QKV_N + 255) / 256)), 256, 0, stream>>>(
      Wqkv, wqkvT, HID, QKV_N);
  convT_bf16_kernel<<<dim3((unsigned)(((size_t)QCOLS * HID + 255) / 256)), 256, 0, stream>>>(
      Wo, woT, QCOLS, HID);

  gate_kernel<<<dim3(T_TOK), 256, 0, stream>>>(hs, gk_w0, gk_w1, gk_b1, g);

  // qkv projection with fused relu/scale epilogue
  gemm_bf16_kernel<1><<<dim3(QKV_N / 128, T_TOK / 128), 256, 32768, stream>>>(
      hsb, wqkvT, bqkv, qkv, T_TOK, QKV_N, HID);

  prep_kernel<<<dim3(NCHUNK, NKVH), 256, 0, stream>>>(qkv, g, expB, qg, kg, kBo, vb);

  dstate_kernel<<<dim3(NCHUNK, NKVH), 256, 32768, stream>>>(kBo, vb, S);

  scan_kernel<<<dim3((NKVH * HD * HD) / 256), 256, 0, stream>>>(S, expB);

  output_kernel<<<dim3(NCHUNK, NHEADS), 256, 81920, stream>>>(qg, kg, vb, S, gnorm, onorm);

  // final o_proj
  gemm_bf16_kernel<0><<<dim3(HID / 128, T_TOK / 128), 256, 32768, stream>>>(
      onorm, woT, nullptr, out, T_TOK, HID, QCOLS);
}
